// GeometricModalityFusion_46059229283222
// MI455X (gfx1250) — matrix operand order repaired and checked by reading, not compiled, not run on hardware
//
#include <hip/hip_runtime.h>
#include <hip/hip_bf16.h>

// ---------------- problem constants ----------------
#define NB 8          // batch
#define NS 2048       // seq
#define ND 1024       // model dim
#define NH 16         // heads
#define NHD 64        // head dim
#define NM 3          // modalities
#define DV 768        // vision in-dim
#define DT 512        // text in-dim
#define DA 1024       // audio in-dim

typedef __bf16 bf16_t;
typedef __bf16 v16bf  __attribute__((ext_vector_type(16)));
typedef __bf16 bf16x8 __attribute__((ext_vector_type(8)));
typedef float  v8f    __attribute__((ext_vector_type(8)));

// ---------------- WMMA GEMM tiling ----------------
#define BM 128
#define BN 128
#define BK 32
#define PAD 8
#define LDB (BK + PAD)

// A fragment 16x32 (MxK) bf16, wave32:
//   lanes 0-15  : row M=lane,    K = {0..7, 16..23}
//   lanes 16-31 : row M=lane-16, K = {8..15, 24..31}
__device__ __forceinline__ v16bf ld_frag_a(const bf16_t* base, int lane) {
  const int m  = lane & 15;
  const int k0 = (lane >> 4) << 3;      // 0 or 8
  const bf16_t* p = base + m * LDB;
  bf16x8 lo = *(const bf16x8*)(p + k0);
  bf16x8 hi = *(const bf16x8*)(p + k0 + 16);
  v16bf r;
#pragma unroll
  for (int i = 0; i < 8; ++i) { r[i] = lo[i]; r[i + 8] = hi[i]; }
  return r;
}

// B fragment 32x16 (KxN) bf16 from n-major LDS tile Bs[n][k]:
//   lanes 0-15  : col N=lane,    K = {0..7, 16..23}
//   lanes 16-31 : col N=lane-16, K = {8..15, 24..31}
__device__ __forceinline__ v16bf ld_frag_b(const bf16_t* base, int lane) {
  const int n  = lane & 15;
  const int k0 = (lane >> 4) << 3;      // 0 or 8
  const bf16_t* p = base + n * LDB + k0;
  bf16x8 lo = *(const bf16x8*)(p);
  bf16x8 hi = *(const bf16x8*)(p + 16);
  v16bf r;
#pragma unroll
  for (int i = 0; i < 8; ++i) { r[i] = lo[i]; r[i + 8] = hi[i]; }
  return r;
}

// Async global->LDS copy of 16 bytes (CDNA5; tracked with ASYNCcnt).
__device__ __forceinline__ void async_copy_b128(void* lds_dst, const void* gsrc) {
  const unsigned lds_off = (unsigned)(unsigned long long)lds_dst;   // low 32 bits = LDS offset
  const unsigned long long ga = (unsigned long long)gsrc;
  asm volatile("global_load_async_to_lds_b128 %0, %1, off"
               :: "v"(lds_off), "v"(ga) : "memory");
}
__device__ __forceinline__ void wait_async_all() {
  asm volatile("s_wait_asynccnt 0" ::: "memory");
}

// C = A[Mr,K] @ Wt[N,K]^T + bias, batched-stride addressing on A and C so the
// same kernel serves feats[:,m] slices, Q/K/V, and the output projection.
// Block tile 128x128, 8 waves, each wave owns a 32x64 sub-tile (2x4 WMMA frags).
__global__ __launch_bounds__(256)
void gemm_bf16_wmma(const bf16_t* __restrict__ A, const bf16_t* __restrict__ Wt,
                    const float* __restrict__ bias,
                    float* __restrict__ Cf, bf16_t* __restrict__ Cb,
                    int Mr, int K, int N,
                    int rpb_a, long long a_bstride,
                    int rpb_c, long long c_bstride) {
  __shared__ __align__(16) bf16_t As[BM][LDB];
  __shared__ __align__(16) bf16_t Bs[BN][LDB];

  const int tid  = threadIdx.x;
  const int lane = tid & 31;
  const int wave = tid >> 5;
  const int wm   = wave >> 1;           // 0..3 : 32-row strip
  const int wn   = wave & 1;            // 0..1 : 64-col strip
  const int row0 = blockIdx.y * BM;
  const int col0 = blockIdx.x * BN;

  const long long batch_a = row0 / rpb_a;
  const bf16_t* Ab = A + batch_a * a_bstride + (long long)(row0 % rpb_a) * K;
  const bf16_t* Bb = Wt + (long long)col0 * K;

  v8f acc[2][4];
  const v8f vzero = {0.f, 0.f, 0.f, 0.f, 0.f, 0.f, 0.f, 0.f};
#pragma unroll
  for (int i = 0; i < 2; ++i)
#pragma unroll
    for (int j = 0; j < 4; ++j) acc[i][j] = vzero;

  // Staging: 128x32 bf16 tile = 512 x 16B chunks; 256 threads x 2 chunks each.
  // chunk c: row = c>>2 (0..127), col = (c&3)*8 (elements).
  const int c0 = tid, c1 = tid + 256;
  const int ar0 = c0 >> 2, ac0 = (c0 & 3) * 8;
  const int ar1 = c1 >> 2, ac1 = (c1 & 3) * 8;

  for (int k0 = 0; k0 < K; k0 += BK) {
    async_copy_b128(&As[ar0][ac0], Ab + (long long)ar0 * K + k0 + ac0);
    async_copy_b128(&As[ar1][ac1], Ab + (long long)ar1 * K + k0 + ac1);
    async_copy_b128(&Bs[ar0][ac0], Bb + (long long)ar0 * K + k0 + ac0);
    async_copy_b128(&Bs[ar1][ac1], Bb + (long long)ar1 * K + k0 + ac1);
    if (k0 + BK < K) {   // speculative prefetch of the next K-slab
      __builtin_prefetch((const void*)(Ab + (long long)ar0 * K + k0 + BK + ac0), 0, 1);
      __builtin_prefetch((const void*)(Bb + (long long)ar0 * K + k0 + BK + ac0), 0, 1);
    }
    wait_async_all();
    __syncthreads();

    v16bf af[2], bfr[4];
    af[0] = ld_frag_a(&As[wm * 32][0],      lane);
    af[1] = ld_frag_a(&As[wm * 32 + 16][0], lane);
#pragma unroll
    for (int j = 0; j < 4; ++j)
      bfr[j] = ld_frag_b(&Bs[wn * 64 + j * 16][0], lane);

#pragma unroll
    for (int i = 0; i < 2; ++i)
#pragma unroll
      for (int j = 0; j < 4; ++j)
        acc[i][j] = __builtin_amdgcn_wmma_f32_16x16x32_bf16(
            false, af[i], false, bfr[j], (short)0, acc[i][j], false, false);
    __syncthreads();
  }

  // epilogue: C/D layout — lane n = lane&15, vgpr r -> M = r + 8*(lane>=16)
  const int nl    = lane & 15;
  const int mbase = (lane >> 4) * 8;
#pragma unroll
  for (int fm = 0; fm < 2; ++fm) {
#pragma unroll
    for (int fn = 0; fn < 4; ++fn) {
      const int gcol = col0 + wn * 64 + fn * 16 + nl;
      const float bv = bias ? bias[gcol] : 0.0f;
#pragma unroll
      for (int r = 0; r < 8; ++r) {
        const int grow = row0 + wm * 32 + fm * 16 + mbase + r;
        const long long cidx =
            (long long)(grow / rpb_c) * c_bstride +
            (long long)(grow % rpb_c) * N + gcol;
        const float v = acc[fm][fn][r] + bv;
        if (Cf) Cf[cidx] = v;
        if (Cb) Cb[cidx] = (bf16_t)v;
      }
    }
  }
}

// ---------------- conversion kernels ----------------
__global__ void cvt_f32_bf16(const float* __restrict__ src, bf16_t* __restrict__ dst, int n) {
  int i = blockIdx.x * blockDim.x + threadIdx.x;
  if (i < n) dst[i] = (bf16_t)src[i];
}

// W[K,N] fp32 -> Wt[N,K] bf16
__global__ void cvt_wT_bf16(const float* __restrict__ W, bf16_t* __restrict__ Wt, int K, int N) {
  int i = blockIdx.x * blockDim.x + threadIdx.x;
  if (i >= K * N) return;
  int k = i / N, n = i % N;
  Wt[(long long)n * K + k] = (bf16_t)W[(long long)k * N + n];
}

__global__ void zero_f32(float* p, int n) {
  int i = blockIdx.x * blockDim.x + threadIdx.x;
  if (i < n) p[i] = 0.0f;
}

// ---------------- per-(b,h,s) cross-modality MHA ----------------
__device__ __forceinline__ float wave_sum(float v) {
#pragma unroll
  for (int off = 16; off; off >>= 1) v += __shfl_xor(v, off, 32);
  return v;
}

__global__ __launch_bounds__(256)
void mha_kernel(const bf16_t* __restrict__ Qb, const bf16_t* __restrict__ Kb,
                const bf16_t* __restrict__ Vb, const float* __restrict__ temp,
                float* __restrict__ mha) {
  const int item = blockIdx.x * 8 + (threadIdx.x >> 5);   // (b,h,s)
  const int lane = threadIdx.x & 31;
  const int b = item / (NH * NS);
  const int hs = item % (NH * NS);
  const int h = hs / NS, s = hs % NS;

  const float t = fabsf(temp[0]);
  const float scale = 1.0f / (8.0f * t);                  // sqrt(64)=8

  const long long qoff = ((long long)b * NS + s) * ND + h * NHD + lane * 2;
  const float q0 = (float)Qb[qoff], q1 = (float)Qb[qoff + 1];

  float sc[NM];
  long long koff[NM];
#pragma unroll
  for (int m = 0; m < NM; ++m) {
    koff[m] = (((long long)(b * NM + m)) * NS + s) * ND + h * NHD + lane * 2;
    float p = q0 * (float)Kb[koff[m]] + q1 * (float)Kb[koff[m] + 1];
    sc[m] = wave_sum(p) * scale;
  }
  const float mx = fmaxf(sc[0], fmaxf(sc[1], sc[2]));
  float e0 = __expf(sc[0] - mx), e1 = __expf(sc[1] - mx), e2 = __expf(sc[2] - mx);
  const float inv = 1.0f / (e0 + e1 + e2);
  e0 *= inv; e1 *= inv; e2 *= inv;

  float o0 = 0.f, o1 = 0.f;
  o0 += e0 * (float)Vb[koff[0]];     o1 += e0 * (float)Vb[koff[0] + 1];
  o0 += e1 * (float)Vb[koff[1]];     o1 += e1 * (float)Vb[koff[1] + 1];
  o0 += e2 * (float)Vb[koff[2]];     o1 += e2 * (float)Vb[koff[2] + 1];
  mha[qoff] = o0; mha[qoff + 1] = o1;
}

// ---------------- per-(b,s) Gram / cosine reduction ----------------
__global__ __launch_bounds__(256)
void gram_kernel(const bf16_t* __restrict__ feats,
                 float* __restrict__ cosAcc, float* __restrict__ Gacc) {
  const int item = blockIdx.x * 8 + (threadIdx.x >> 5);   // b*NS + s
  const int lane = threadIdx.x & 31;
  const int b = item / NS, s = item % NS;

  const bf16_t* f0 = feats + (((long long)(b * NM + 0)) * NS + s) * ND;
  const bf16_t* f1 = feats + (((long long)(b * NM + 1)) * NS + s) * ND;
  const bf16_t* f2 = feats + (((long long)(b * NM + 2)) * NS + s) * ND;

  float s00 = 0, s01 = 0, s02 = 0, s11 = 0, s12 = 0, s22 = 0;
  for (int d = lane; d < ND; d += 32) {
    const float a = (float)f0[d], c = (float)f1[d], e = (float)f2[d];
    s00 += a * a; s01 += a * c; s02 += a * e;
    s11 += c * c; s12 += c * e; s22 += e * e;
  }
  s00 = wave_sum(s00); s01 = wave_sum(s01); s02 = wave_sum(s02);
  s11 = wave_sum(s11); s12 = wave_sum(s12); s22 = wave_sum(s22);

  if (lane == 0) {
    const float n0 = fmaxf(sqrtf(s00), 1e-12f);
    const float n1 = fmaxf(sqrtf(s11), 1e-12f);
    const float n2 = fmaxf(sqrtf(s22), 1e-12f);
    const float invS = 1.0f / (float)NS;
    float* cb = cosAcc + b * 9;
    float* gb = Gacc + b * 9;
    atomicAdd(&cb[0], (s00 / (n0 * n0)) * invS);
    atomicAdd(&cb[4], (s11 / (n1 * n1)) * invS);
    atomicAdd(&cb[8], (s22 / (n2 * n2)) * invS);
    const float c01 = (s01 / (n0 * n1)) * invS;
    const float c02 = (s02 / (n0 * n2)) * invS;
    const float c12 = (s12 / (n1 * n2)) * invS;
    atomicAdd(&cb[1], c01); atomicAdd(&cb[3], c01);
    atomicAdd(&cb[2], c02); atomicAdd(&cb[6], c02);
    atomicAdd(&cb[5], c12); atomicAdd(&cb[7], c12);
    atomicAdd(&gb[0], s00); atomicAdd(&gb[4], s11); atomicAdd(&gb[8], s22);
    atomicAdd(&gb[1], s01); atomicAdd(&gb[3], s01);
    atomicAdd(&gb[2], s02); atomicAdd(&gb[6], s02);
    atomicAdd(&gb[5], s12); atomicAdd(&gb[7], s12);
  }
}

// ---------------- per-batch angular + Cayley weights ----------------
__global__ void weights_kernel(const float* __restrict__ cosAcc,
                               const float* __restrict__ Gacc,
                               const float* __restrict__ temp,
                               const float* __restrict__ attn_w,
                               float* __restrict__ wcoef) {
  const int b = threadIdx.x;
  if (b >= NB) return;
  const float t = fabsf(temp[0]);

  // fusion weights softmax
  float fm = fmaxf(attn_w[0], fmaxf(attn_w[1], attn_w[2]));
  float f0 = __expf(attn_w[0] - fm), f1 = __expf(attn_w[1] - fm), f2 = __expf(attn_w[2] - fm);
  const float finv = 1.0f / (f0 + f1 + f2);
  f0 *= finv; f1 *= finv; f2 *= finv;

  // angular attention
  const float rw[3] = {1.0f, -0.75f, 0.75f};
  float logit[3];
  for (int i = 0; i < 3; ++i) {
    float acc = 0.f;
    for (int j = 0; j < 3; ++j) {
      if (i == j) continue;
      float c = cosAcc[b * 9 + i * 3 + j];
      c = fminf(fmaxf(c, -1.0f + 1e-7f), 1.0f - 1e-7f);
      acc += __expf(-acosf(c) / t) * rw[j];
    }
    logit[i] = acc;
  }
  float lm = fmaxf(logit[0], fmaxf(logit[1], logit[2]));
  float a0 = __expf(logit[0] - lm), a1 = __expf(logit[1] - lm), a2 = __expf(logit[2] - lm);
  const float ainv = 1.0f / (a0 + a1 + a2);
  float aw[3] = {a0 * ainv, a1 * ainv, a2 * ainv};

  // Cayley-Menger via simplex Gram
  float A[3][3][3];
  for (int i = 0; i < 3; ++i)
    for (int p = 0; p < 3; ++p)
      for (int k = 0; k < 3; ++k) A[i][p][k] = 0.f;
  const float PI = 3.14159265358979323846f;
  for (int i = 0; i < 3; ++i) {
    A[i][0][i] = 1.0f;
    for (int j = 0; j < 2; ++j) {
      const float a = (j + 1) * PI * 0.25f;
      A[i][j + 1][i] += __cosf(a);
      A[i][j + 1][(i + j + 1) % 3] += __sinf(a);
    }
  }
  const float* Gb = Gacc + b * 9;
  float vol[3];
  for (int i = 0; i < 3; ++i) {
    float Gi[3][3];
    for (int p = 0; p < 3; ++p)
      for (int q = 0; q < 3; ++q) {
        float acc = 0.f;
        for (int k = 0; k < 3; ++k)
          for (int l = 0; l < 3; ++l)
            acc += A[i][p][k] * Gb[k * 3 + l] * A[i][q][l];
        Gi[p][q] = acc;
      }
    float v = 0.f;
    for (int p = 0; p < 3; ++p)
      for (int q = 0; q < 3; ++q)
        v += Gi[p][p] + Gi[q][q] - 2.0f * Gi[p][q];
    vol[i] = v / 9.0f;
  }
  float vm = fmaxf(vol[0] / t, fmaxf(vol[1] / t, vol[2] / t));
  float c0 = __expf(vol[0] / t - vm), c1 = __expf(vol[1] / t - vm), c2 = __expf(vol[2] / t - vm);
  const float cinv = 1.0f / (c0 + c1 + c2);
  float cw[3] = {c0 * cinv, c1 * cinv, c2 * cinv};

  for (int i = 0; i < 3; ++i) wcoef[b * 4 + i] = f1 * aw[i] + f2 * cw[i];
  wcoef[b * 4 + 3] = f0;
}

// ---------------- fusion: fw0*mha + sum_i coef_i*feats_i -> bf16 ----------------
__global__ __launch_bounds__(256)
void fuse_kernel(const float* __restrict__ mha, const bf16_t* __restrict__ feats,
                 const float* __restrict__ wcoef, bf16_t* __restrict__ fused) {
  const long long idx = (long long)blockIdx.x * blockDim.x + threadIdx.x;
  const int b = (int)(idx / ((long long)NS * ND));
  const long long rem = idx % ((long long)NS * ND);
  float v = wcoef[b * 4 + 3] * mha[idx];
#pragma unroll
  for (int m = 0; m < NM; ++m)
    v += wcoef[b * 4 + m] * (float)feats[((long long)(b * NM + m)) * NS * ND + rem];
  fused[idx] = (bf16_t)v;
}

// ---------------- host-side launch ----------------
static inline void launch_gemm(const bf16_t* A, const bf16_t* Wt, const float* bias,
                               float* Cf, bf16_t* Cb, int Mr, int K, int N,
                               int rpb_a, long long a_bs, int rpb_c, long long c_bs,
                               hipStream_t s) {
  dim3 grid(N / BN, Mr / BM);
  gemm_bf16_wmma<<<grid, dim3(256), 0, s>>>(A, Wt, bias, Cf, Cb, Mr, K, N,
                                            rpb_a, a_bs, rpb_c, c_bs);
}

extern "C" void kernel_launch(void* const* d_in, const int* in_sizes, int n_in,
                              void* d_out, int out_size, void* d_ws, size_t ws_size,
                              hipStream_t stream) {
  const float* vision   = (const float*)d_in[0];
  const float* w_vision = (const float*)d_in[1];
  const float* b_vision = (const float*)d_in[2];
  const float* text     = (const float*)d_in[3];
  const float* w_text   = (const float*)d_in[4];
  const float* b_text   = (const float*)d_in[5];
  const float* audio    = (const float*)d_in[6];
  const float* w_audio  = (const float*)d_in[7];
  const float* b_audio  = (const float*)d_in[8];
  const float* wq = (const float*)d_in[9];   const float* bq = (const float*)d_in[10];
  const float* wk = (const float*)d_in[11];  const float* bk = (const float*)d_in[12];
  const float* wv = (const float*)d_in[13];  const float* bv = (const float*)d_in[14];
  const float* wo = (const float*)d_in[15];  const float* bo = (const float*)d_in[16];
  const float* temperature  = (const float*)d_in[17];
  const float* attn_weights = (const float*)d_in[18];
  float* out = (float*)d_out;

  char* ws = (char*)d_ws;
  size_t off = 0;
  auto alloc = [&](size_t bytes) -> void* {
    void* p = ws + off;
    off = (off + bytes + 255) & ~(size_t)255;
    return p;
  };

  const long long BS = (long long)NB * NS;            // 16384
  const long long BMS = BS * NM;                      // 49152
  bf16_t* vis_b   = (bf16_t*)alloc(BS * DV * 2);
  bf16_t* txt_b   = (bf16_t*)alloc(BS * DT * 2);
  bf16_t* aud_b   = (bf16_t*)alloc(BS * DA * 2);
  bf16_t* wvis_t  = (bf16_t*)alloc((size_t)DV * ND * 2);
  bf16_t* wtxt_t  = (bf16_t*)alloc((size_t)DT * ND * 2);
  bf16_t* waud_t  = (bf16_t*)alloc((size_t)DA * ND * 2);
  bf16_t* wq_t    = (bf16_t*)alloc((size_t)ND * ND * 2);
  bf16_t* wk_t    = (bf16_t*)alloc((size_t)ND * ND * 2);
  bf16_t* wv_t    = (bf16_t*)alloc((size_t)ND * ND * 2);
  bf16_t* wo_t    = (bf16_t*)alloc((size_t)ND * ND * 2);
  bf16_t* feats_b = (bf16_t*)alloc(BMS * ND * 2);
  bf16_t* Qb      = (bf16_t*)alloc(BS * ND * 2);
  bf16_t* Kb      = (bf16_t*)alloc(BMS * ND * 2);
  bf16_t* Vb      = (bf16_t*)alloc(BMS * ND * 2);
  float*  mha_f   = (float*) alloc(BS * ND * 4);
  bf16_t* fused_b = (bf16_t*)alloc(BS * ND * 2);
  float*  cosAcc  = (float*) alloc(NB * 9 * 4);
  float*  Gacc    = (float*) alloc(NB * 9 * 4);
  float*  wcoef   = (float*) alloc(NB * 4 * 4);

  // 1) conversions
  {
    int n;
    n = (int)(BS * DV); cvt_f32_bf16<<<(n + 255) / 256, 256, 0, stream>>>(vision, vis_b, n);
    n = (int)(BS * DT); cvt_f32_bf16<<<(n + 255) / 256, 256, 0, stream>>>(text, txt_b, n);
    n = (int)(BS * DA); cvt_f32_bf16<<<(n + 255) / 256, 256, 0, stream>>>(audio, aud_b, n);
    n = DV * ND; cvt_wT_bf16<<<(n + 255) / 256, 256, 0, stream>>>(w_vision, wvis_t, DV, ND);
    n = DT * ND; cvt_wT_bf16<<<(n + 255) / 256, 256, 0, stream>>>(w_text,   wtxt_t, DT, ND);
    n = DA * ND; cvt_wT_bf16<<<(n + 255) / 256, 256, 0, stream>>>(w_audio,  waud_t, DA, ND);
    n = ND * ND; cvt_wT_bf16<<<(n + 255) / 256, 256, 0, stream>>>(wq, wq_t, ND, ND);
    n = ND * ND; cvt_wT_bf16<<<(n + 255) / 256, 256, 0, stream>>>(wk, wk_t, ND, ND);
    n = ND * ND; cvt_wT_bf16<<<(n + 255) / 256, 256, 0, stream>>>(wv, wv_t, ND, ND);
    n = ND * ND; cvt_wT_bf16<<<(n + 255) / 256, 256, 0, stream>>>(wo, wo_t, ND, ND);
    zero_f32<<<1, 256, 0, stream>>>(cosAcc, NB * 9);
    zero_f32<<<1, 256, 0, stream>>>(Gacc, NB * 9);
  }

  const long long feat_bs = (long long)NM * NS * ND;  // batch stride inside feats

  // 2) modality projections -> feats[b][m][s][d] (bf16)
  launch_gemm(vis_b, wvis_t, b_vision, nullptr, feats_b + 0LL * NS * ND,
              (int)BS, DV, ND, (int)BS, 0, NS, feat_bs, stream);
  launch_gemm(txt_b, wtxt_t, b_text, nullptr, feats_b + 1LL * NS * ND,
              (int)BS, DT, ND, (int)BS, 0, NS, feat_bs, stream);
  launch_gemm(aud_b, waud_t, b_audio, nullptr, feats_b + 2LL * NS * ND,
              (int)BS, DA, ND, (int)BS, 0, NS, feat_bs, stream);

  // 3) Q (from feats[:,0] slice), K, V
  launch_gemm(feats_b, wq_t, bq, nullptr, Qb,
              (int)BS, ND, ND, NS, feat_bs, (int)BS, 0, stream);
  launch_gemm(feats_b, wk_t, bk, nullptr, Kb,
              (int)BMS, ND, ND, (int)BMS, 0, (int)BMS, 0, stream);
  launch_gemm(feats_b, wv_t, bv, nullptr, Vb,
              (int)BMS, ND, ND, (int)BMS, 0, (int)BMS, 0, stream);

  // 4) cross-modality MHA (softmax over M per (b,h,s))
  mha_kernel<<<(NB * NH * NS) / 8, 256, 0, stream>>>(Qb, Kb, Vb, temperature, mha_f);

  // 5) Gram + cosine reductions
  gram_kernel<<<(NB * NS) / 8, 256, 0, stream>>>(feats_b, cosAcc, Gacc);

  // 6) per-batch angular / Cayley-Menger weights
  weights_kernel<<<1, 32, 0, stream>>>(cosAcc, Gacc, temperature, attn_weights, wcoef);

  // 7) fusion -> bf16
  fuse_kernel<<<(int)((BS * ND) / 256), 256, 0, stream>>>(mha_f, feats_b, wcoef, fused_b);

  // 8) output projection -> fp32 d_out
  launch_gemm(fused_b, wo_t, bo, out, nullptr,
              (int)BS, ND, ND, (int)BS, 0, (int)BS, 0, stream);

  (void)in_sizes; (void)n_in; (void)out_size; (void)ws_size;
}
